// Histogram_11184094839163
// MI455X (gfx1250) — compile-verified, matrix operand order checked by reading, not run on hardware
//
#include <hip/hip_runtime.h>
#include <hip/hip_bf16.h>

typedef __attribute__((ext_vector_type(16))) _Float16 v16h;
typedef __attribute__((ext_vector_type(8)))  float    v8f;

#define NUM_BINS 256
#define HW (128 * 128)
#define NC 24
#define WAVES_PER_BLOCK 8
#define PIX_PER_WAVE (HW / WAVES_PER_BLOCK)   // 2048
#define ITERS (PIX_PER_WAVE / 32)             // 64 WMMA iterations per wave

// Soft histogram via sigmoid kernel, reduced with v_wmma_f32_16x16x32_f16.
// A = ones(16x32) f16, B[k, n] = w(pixel k, bin n) f16, C accumulates f32.
// D[m,n] = sum_k B[k,n]  -> per-bin partial sums, invariant to K layout.
//
// Math: with z1 = (x - c + d/2)/b, z2 = z1 - d/b (constant shift),
//   w = sigmoid(z1) - sigmoid(z2) = e*(C-1) / ((1+e)*(1+C*e)),
// where e = exp(-z1), C = exp(d/b) = exp(3.90625).
// -> ONE v_exp_f32 + ONE v_rcp_f32 per (pixel,bin) pair.
__global__ __launch_bounds__(256) void Histogram_soft_wmma_kernel(
    const float* __restrict__ x, float* __restrict__ out) {
  const int block = blockIdx.x;
  const int ch = block >> 4;          // 0..23 : (N*C) channel
  const int bt = block & 15;          // bin tile: 16 bins per tile
  const int tid  = threadIdx.x;
  const int wave = tid >> 5;
  const int lane = tid & 31;
  const int n    = lane & 15;                 // B-matrix column = bin within tile
  const int koff2 = (lane >> 4) << 3;         // float2 offset: lanes 0-15 -> px 0..15, 16-31 -> px 16..31

  // Constants (delta = 1/256, bandwidth = 1e-3):
  //   k   = log2(e)/bandwidth               = 1442.69504
  //   cA  = bin * delta * k                 = bin * 5.63552750  (since center - delta/2 = bin*delta)
  //   C   = exp(delta/bandwidth) = exp(3.90625) = 49.712185
  const float kscale = 1442.6950408889634f;
  const float Cbig   = 49.712185f;
  const float Cm1    = 48.712185f;
  const int   bin    = bt * 16 + n;
  const float cA     = (float)bin * 5.6355275034725f;

  const float2* __restrict__ px = (const float2*)(x + ch * HW);
  int p2 = (wave * PIX_PER_WAVE) >> 1;        // float2 index of this wave's pixel window

  v16h a;
  #pragma unroll
  for (int i = 0; i < 16; ++i) a[i] = (_Float16)1.0f;

  v8f c = {};

  for (int it = 0; it < ITERS; ++it) {
    v16h b;
    #pragma unroll
    for (int j = 0; j < 8; ++j) {
      float2 xv = px[p2 + koff2 + j];
      // arg = -z1*log2(e); clamp to 100 so e <= 2^100 keeps num finite; p may overflow to inf, rcp -> 0 (no inf*0).
      float arg0 = fminf(__builtin_fmaf(xv.x, -kscale, cA), 100.0f);
      float e0   = __builtin_amdgcn_exp2f(arg0);
      float p0   = (1.0f + e0) * (1.0f + Cbig * e0);
      float w0   = (e0 * Cm1) * __builtin_amdgcn_rcpf(p0);

      float arg1 = fminf(__builtin_fmaf(xv.y, -kscale, cA), 100.0f);
      float e1   = __builtin_amdgcn_exp2f(arg1);
      float p1   = (1.0f + e1) * (1.0f + Cbig * e1);
      float w1   = (e1 * Cm1) * __builtin_amdgcn_rcpf(p1);

      b[2 * j]     = (_Float16)w0;
      b[2 * j + 1] = (_Float16)w1;
    }
    // 8 args: (neg_a, A, neg_b, B, c_mod, C, reuse_a, reuse_b)
    c = __builtin_amdgcn_wmma_f32_16x16x32_f16(
        false, a, false, b, (short)0, c, false, false);
    p2 += 16;  // advance 32 pixels
  }

  // Every D element in column N equals the column sum; VGPR0, lanes 0-15 = (M=0, N=lane).
  __shared__ float partial[WAVES_PER_BLOCK][16];
  if (lane < 16) partial[wave][lane] = c[0];
  __syncthreads();
  if (tid < 16) {
    float s = 0.0f;
    #pragma unroll
    for (int w = 0; w < WAVES_PER_BLOCK; ++w) s += partial[w][tid];
    out[ch * NUM_BINS + bt * 16 + tid] = s;
  }
}

extern "C" void kernel_launch(void* const* d_in, const int* in_sizes, int n_in,
                              void* d_out, int out_size, void* d_ws, size_t ws_size,
                              hipStream_t stream) {
  const float* x = (const float*)d_in[0];   // [8, 3, 128, 128] f32
  float* out = (float*)d_out;               // [8, 3, 256] f32
  (void)in_sizes; (void)n_in; (void)out_size; (void)d_ws; (void)ws_size;
  dim3 grid(NC * 16);   // 384 blocks: (channel, bin-tile)
  dim3 blk(256);        // 8 waves; each wave reduces 2048 pixels
  Histogram_soft_wmma_kernel<<<grid, blk, 0, stream>>>(x, out);
}
